// SafeRocketLeagueGCN_38594576122353
// MI455X (gfx1250) — matrix operand order repaired and checked by reading, not compiled, not run on hardware
//
#include <hip/hip_runtime.h>
#include <math.h>

#define NGRAPH 131072
#define GPB 32                 // graphs per workgroup
#define NODES_PB (GPB * 6)     // 192 nodes per workgroup

// ---- workspace byte offsets (written by prep_kernel, read by gcn_kernel) ----
#define WS_SCAL 0              // 512 floats of folded BN scale/shift (2048 B)
#define WS_W1P  2048           // conv1 weight, WMMA B-operand swizzled, 1024 bf16
#define WS_W2P  4096           // conv2 weight, 1024 bf16
#define WS_SW1P 6144           // MLP1 weight 64x128 (K padded 46->64), 8192 bf16
#define WS_SW2P 22528          // MLP2 weight 128x64, 8192 bf16

// float indices inside WS_SCAL block
#define S1 0
#define H1 32
#define S2 64
#define H2 96
#define S3 128
#define H3 256
#define S4 384
#define H4 448

// ---- LDS byte offsets (overlapped lifetimes) ----
// [0..12288)      xbf (phases 0-1) / hbf (2-3) / zbf(0..4096)+z2bf(4096..12288) (4-6)
// [12288..16128)  edge weights, async-staged (phase 0 only)
// [16128..20736)  Mn 6x6 norm matrices (phases 0-4)
// [20736..45312)  T f32 GEMM tiles (1-4) / z3 f32 (6-7, first 8192 B)
// [45312..47104)  global features, async-staged (read in phase 4)
#define SM_BYTES 47104

typedef unsigned short u16;
typedef __attribute__((ext_vector_type(16))) __bf16 v16bf;
typedef __attribute__((ext_vector_type(8)))  float  v8f;

__device__ __forceinline__ u16 f2bf(float f) {
    union { float f; unsigned int u; } c; c.f = f;
    unsigned int r = c.u + 0x7FFFu + ((c.u >> 16) & 1u);   // round-to-nearest-even
    return (u16)(r >> 16);
}

// Async 16B global->LDS copy (CDNA5 GLOBAL_LOAD_ASYNC_TO_LDS_B128, ASYNCcnt).
// VDST carries the LDS byte address; generic-pointer low 32 bits are the LDS
// offset per the flat-aperture mapping.
__device__ __forceinline__ void async_copy16(void* lds, const void* gptr) {
    unsigned int laddr = (unsigned int)(unsigned long long)lds;
    asm volatile("global_load_async_to_lds_b128 %0, %1, off"
                 :: "v"(laddr), "v"(gptr) : "memory");
}
__device__ __forceinline__ void async_wait0() {
    asm volatile("s_wait_asynccnt 0x0" ::: "memory");
}

// Pack WMMA A operand (bf16 16x32) from an LDS row of >=32 bf16.
// Layout: lane half 0 holds K {0..15}, half 1 holds K {16..31},
// stored as consecutive K-pairs per 32-bit VGPR.
__device__ __forceinline__ v16bf packA(const u16* base, int hh) {
    union { v16bf v; unsigned int u[8]; } a;
    const u16* p = base + hh * 16;
    a.u[0] = *(const unsigned int*)(p + 0);
    a.u[1] = *(const unsigned int*)(p + 2);
    a.u[2] = *(const unsigned int*)(p + 4);
    a.u[3] = *(const unsigned int*)(p + 6);
    a.u[4] = *(const unsigned int*)(p + 8);
    a.u[5] = *(const unsigned int*)(p + 10);
    a.u[6] = *(const unsigned int*)(p + 12);
    a.u[7] = *(const unsigned int*)(p + 14);
    return a.v;
}

__device__ __forceinline__ v8f wmma_bf16(v16bf a, v16bf b, v8f c) {
    return __builtin_amdgcn_wmma_f32_16x16x32_bf16(false, a, false, b, (short)0, c,
                                                   false, false);
}

// one 16-row node-transform tile: [16 x 32] bf16 x [32 x 32] weight -> T f32
__device__ __forceinline__ void gemm_tile(const u16* __restrict__ Abuf,
                                          const u16* __restrict__ WP,
                                          float* __restrict__ T,
                                          int lane, int t) {
    const int m  = lane & 15;
    const int hh = lane >> 4;
    v16bf a = packA(&Abuf[(t * 16 + m) * 32], hh);
    #pragma unroll
    for (int nt = 0; nt < 2; ++nt) {
        v16bf b = *(const v16bf*)(const void*)(WP + nt * 512 + lane * 16);
        v8f acc = {0.f, 0.f, 0.f, 0.f, 0.f, 0.f, 0.f, 0.f};
        acc = wmma_bf16(a, b, acc);
        #pragma unroll
        for (int r = 0; r < 8; ++r)
            T[(t * 16 + r + 8 * hh) * 32 + nt * 16 + m] = acc[r];
    }
}

// ---------------------------------------------------------------------------
// Prep: fold BN into scale/shift, swizzle weights into WMMA B-operand order.
// B operand (32x16 bf16): lane L covers column N=L%16, K = (L/16)*16 + e,
// e = 0..15 stored contiguously per lane (32 bytes -> one vector load).
// ---------------------------------------------------------------------------
__global__ __launch_bounds__(256) void prep_kernel(
    const float* __restrict__ w1,  const float* __restrict__ b1,
    const float* __restrict__ g1,  const float* __restrict__ be1,
    const float* __restrict__ m1,  const float* __restrict__ v1,
    const float* __restrict__ w2,  const float* __restrict__ b2,
    const float* __restrict__ g2,  const float* __restrict__ be2,
    const float* __restrict__ m2,  const float* __restrict__ v2,
    const float* __restrict__ sw1, const float* __restrict__ sb1,
    const float* __restrict__ sg1, const float* __restrict__ sbe1,
    const float* __restrict__ sm1, const float* __restrict__ sv1,
    const float* __restrict__ sw2, const float* __restrict__ sb2,
    const float* __restrict__ sg2, const float* __restrict__ sbe2,
    const float* __restrict__ sm2, const float* __restrict__ sv2,
    unsigned char* __restrict__ ws) {
    const int tid = threadIdx.x;
    float* S = (float*)(ws + WS_SCAL);
    for (int i = tid; i < 32; i += 256) {
        float s = g1[i] * rsqrtf(v1[i] + 1e-5f);
        S[S1 + i] = s; S[H1 + i] = (b1[i] - m1[i]) * s + be1[i];
    }
    for (int i = tid; i < 32; i += 256) {
        float s = g2[i] * rsqrtf(v2[i] + 1e-5f);
        S[S2 + i] = s; S[H2 + i] = (b2[i] - m2[i]) * s + be2[i];
    }
    for (int i = tid; i < 128; i += 256) {
        float s = sg1[i] * rsqrtf(sv1[i] + 1e-5f);
        S[S3 + i] = s; S[H3 + i] = (sb1[i] - sm1[i]) * s + sbe1[i];
    }
    for (int i = tid; i < 64; i += 256) {
        float s = sg2[i] * rsqrtf(sv2[i] + 1e-5f);
        S[S4 + i] = s; S[H4 + i] = (sb2[i] - sm2[i]) * s + sbe2[i];
    }
    u16* W1P = (u16*)(ws + WS_W1P);
    for (int i = tid; i < 1024; i += 256) {
        int nt = i >> 9, r = i & 511, L = r >> 4, e = r & 15;
        int K = ((L >= 16) ? 16 : 0) + e, N = nt * 16 + (L & 15);
        W1P[i] = (K < 13) ? f2bf(w1[K * 32 + N]) : (u16)0;     // pad K 13->32
    }
    u16* W2P = (u16*)(ws + WS_W2P);
    for (int i = tid; i < 1024; i += 256) {
        int nt = i >> 9, r = i & 511, L = r >> 4, e = r & 15;
        int K = ((L >= 16) ? 16 : 0) + e, N = nt * 16 + (L & 15);
        W2P[i] = f2bf(w2[K * 32 + N]);
    }
    u16* P1 = (u16*)(ws + WS_SW1P);                            // 46x128 -> 64x128
    for (int i = tid; i < 8192; i += 256) {
        int t = i >> 9, r = i & 511, L = r >> 4, e = r & 15;
        int nt = t >> 1, kc = t & 1;
        int K = kc * 32 + ((L >= 16) ? 16 : 0) + e, N = nt * 16 + (L & 15);
        P1[i] = (K < 46) ? f2bf(sw1[K * 128 + N]) : (u16)0;
    }
    u16* P2 = (u16*)(ws + WS_SW2P);                            // 128x64
    for (int i = tid; i < 8192; i += 256) {
        int t = i >> 9, r = i & 511, L = r >> 4, e = r & 15;
        int nt = t >> 2, kc = t & 3;
        int K = kc * 32 + ((L >= 16) ? 16 : 0) + e, N = nt * 16 + (L & 15);
        P2[i] = f2bf(sw2[K * 64 + N]);
    }
}

// ---------------------------------------------------------------------------
// Fused GCN + pool + MLP + heads. One workgroup (8 wave32) per 32 graphs.
// ---------------------------------------------------------------------------
__global__ __launch_bounds__(256) void gcn_kernel(
    const float* __restrict__ xg,  const float* __restrict__ ewg,
    const float* __restrict__ gfg,
    const float* __restrict__ wo,  const float* __restrict__ bo,
    const float* __restrict__ wb,  const float* __restrict__ bb,
    const unsigned char* __restrict__ ws, float* __restrict__ out) {
    const int tid  = threadIdx.x;
    const int lane = tid & 31;
    const int wv   = tid >> 5;
    const int g0   = blockIdx.x * GPB;
    const int n0   = g0 * 6;

    __shared__ __align__(16) unsigned char smem[SM_BYTES];
    u16*   xbf  = (u16*)smem;                    // [192][32] bf16
    u16*   hbf  = (u16*)smem;                    // reuses xbf region
    u16*   zbf  = (u16*)smem;                    // [32][64]  (after phase 3)
    u16*   z2bf = (u16*)(smem + 4096);           // [32][128]
    float* ewS  = (float*)(smem + 12288);        // [960]
    float* Mn   = (float*)(smem + 16128);        // [32][6][6]
    float* T    = (float*)(smem + 20736);        // [192][32] f32
    float* z3   = (float*)(smem + 20736);        // [32][64]  (after phase 4)
    float* gfS  = (float*)(smem + 45312);        // [32][14]

    const float* S   = (const float*)(ws + WS_SCAL);
    const u16* W1P   = (const u16*)(ws + WS_W1P);
    const u16* W2P   = (const u16*)(ws + WS_W2P);
    const u16* SW1P  = (const u16*)(ws + WS_SW1P);
    const u16* SW2P  = (const u16*)(ws + WS_SW2P);

    // ---- phase 0: async-stage edge weights + global features into LDS ----
    if (tid < 240) async_copy16(&ewS[tid * 4], ewg + g0 * 30 + tid * 4);   // 3840 B
    if (tid < 112) async_copy16(&gfS[tid * 4], gfg + g0 * 14 + tid * 4);   // 1792 B

    // ---- phase 0: stage x as bf16 (K-padded 13->32) ----
    for (int i = tid; i < 192 * 13; i += 256) {
        float v = xg[n0 * 13 + i];
        int r = i / 13, k = i - 13 * r;
        xbf[r * 32 + k] = f2bf(v);
    }
    for (int i = tid; i < 192 * 19; i += 256) {
        int r = i / 19, k = 13 + (i - 19 * r);
        xbf[r * 32 + k] = 0;
    }
    async_wait0();
    __syncthreads();

    // ---- phase 0b: per-graph normalized adjacency (topology is static) ----
    if (tid < GPB) {
        const float* e = &ewS[tid * 30];
        float deg[6] = {1.f, 1.f, 1.f, 1.f, 1.f, 1.f};      // self-loop weight 1
        #pragma unroll
        for (int k = 0; k < 30; ++k) {
            int s = k / 5, j = k - 5 * s, d = j + (j >= s ? 1 : 0);
            deg[d] += e[k];
        }
        float di[6];
        #pragma unroll
        for (int c = 0; c < 6; ++c) di[c] = rsqrtf(deg[c]);
        float* M = &Mn[tid * 36];
        #pragma unroll
        for (int c = 0; c < 6; ++c)
            #pragma unroll
            for (int r = 0; r < 6; ++r) {
                if (r == c) M[c * 6 + r] = di[c] * di[c];
                else {
                    int j = c - (c > r ? 1 : 0);
                    M[c * 6 + r] = di[r] * e[r * 5 + j] * di[c];
                }
            }
    }
    __syncthreads();

    // ---- phase 1: conv1 node transform (WMMA bf16); 12 tiles, waves 0-3 take two ----
    gemm_tile(xbf, W1P, T, lane, wv);
    if (wv < 4) gemm_tile(xbf, W1P, T, lane, wv + 8);
    __syncthreads();

    // ---- phase 2: aggregate + BN1 + ReLU -> hbf ----
    for (int i = tid; i < NODES_PB * 32; i += 256) {
        int n = i >> 5, f = i & 31;
        int gl = n / 6, c = n - 6 * gl;
        const float* M = &Mn[gl * 36 + c * 6];
        const float* Tr = &T[gl * 6 * 32 + f];
        float a = 0.f;
        #pragma unroll
        for (int r = 0; r < 6; ++r) a += M[r] * Tr[r * 32];
        a = fmaxf(a * S[S1 + f] + S[H1 + f], 0.f);
        hbf[n * 32 + f] = f2bf(a);
    }
    __syncthreads();

    // ---- phase 3: conv2 node transform (WMMA bf16) ----
    gemm_tile(hbf, W2P, T, lane, wv);
    if (wv < 4) gemm_tile(hbf, W2P, T, lane, wv + 8);
    __syncthreads();

    // ---- phase 4: aggregate + BN2 + ReLU + mean-pool; build z = [pool|gf|0] ----
    for (int i = tid; i < GPB * 32; i += 256) {
        int gl = i >> 5, f = i & 31;
        const float* M = &Mn[gl * 36];
        float sum = 0.f;
        #pragma unroll
        for (int c = 0; c < 6; ++c) {
            float a = 0.f;
            #pragma unroll
            for (int r = 0; r < 6; ++r) a += M[c * 6 + r] * T[(gl * 6 + r) * 32 + f];
            sum += fmaxf(a * S[S2 + f] + S[H2 + f], 0.f);
        }
        zbf[gl * 64 + f] = f2bf(sum * (1.f / 6.f));
    }
    for (int i = tid; i < GPB * 32; i += 256) {
        int gl = i >> 5, j = i & 31;
        zbf[gl * 64 + 32 + j] = (j < 14) ? f2bf(gfS[gl * 14 + j]) : (u16)0;
    }
    __syncthreads();

    // ---- phase 5: MLP layer 1 (46->128, K padded to 64), WMMA on all 8 waves ----
    // 16 jobs = 2 row-tiles x 8 N-tiles; wave w -> jobs {w, w+8} (same row parity).
    {
        const int m = lane & 15, hh = lane >> 4;
        const int rt = wv & 1;                   // row tile (graphs rt*16..rt*16+15)
        const int row = rt * 16 + m;
        v16bf a0 = packA(&zbf[row * 64 + 0],  hh);
        v16bf a1 = packA(&zbf[row * 64 + 32], hh);
        #pragma unroll
        for (int jj = 0; jj < 2; ++jj) {
            const int nt = (wv >> 1) + jj * 4;
            v8f acc = {0.f, 0.f, 0.f, 0.f, 0.f, 0.f, 0.f, 0.f};
            v16bf b0 = *(const v16bf*)(const void*)(SW1P + (nt * 2 + 0) * 512 + lane * 16);
            acc = wmma_bf16(a0, b0, acc);
            v16bf b1 = *(const v16bf*)(const void*)(SW1P + (nt * 2 + 1) * 512 + lane * 16);
            acc = wmma_bf16(a1, b1, acc);
            int f = nt * 16 + m;
            float sc = S[S3 + f], sh = S[H3 + f];
            #pragma unroll
            for (int r = 0; r < 8; ++r)
                z2bf[(rt * 16 + r + 8 * hh) * 128 + f] = f2bf(fmaxf(acc[r] * sc + sh, 0.f));
        }
    }
    __syncthreads();

    // ---- phase 6: MLP layer 2 (128->64), 8 jobs = 2 row-tiles x 4 N-tiles ----
    {
        const int m = lane & 15, hh = lane >> 4;
        const int rt = wv & 1;
        const int nt = wv >> 1;
        const int row = rt * 16 + m;
        v16bf a[4];
        #pragma unroll
        for (int kc = 0; kc < 4; ++kc) a[kc] = packA(&z2bf[row * 128 + kc * 32], hh);
        v8f acc = {0.f, 0.f, 0.f, 0.f, 0.f, 0.f, 0.f, 0.f};
        #pragma unroll
        for (int kc = 0; kc < 4; ++kc) {
            v16bf b = *(const v16bf*)(const void*)(SW2P + (nt * 4 + kc) * 512 + lane * 16);
            acc = wmma_bf16(a[kc], b, acc);
        }
        int f = nt * 16 + m;
        float sc = S[S4 + f], sh = S[H4 + f];
        #pragma unroll
        for (int r = 0; r < 8; ++r)
            z3[(rt * 16 + r + 8 * hh) * 64 + f] = fmaxf(acc[r] * sc + sh, 0.f);
    }
    __syncthreads();

    // ---- phase 7: sigmoid heads; 64 jobs = 32 graphs x 2 heads ----
    if (tid < 2 * GPB) {
        const int gl = tid >> 1, head = tid & 1;
        const float* hw = head ? wb : wo;
        const float  hb = head ? bb[0] : bo[0];
        const float* zr = &z3[gl * 64];
        float acc = hb;
        #pragma unroll 8
        for (int f = 0; f < 64; ++f) acc += zr[f] * hw[f];
        out[head * NGRAPH + g0 + gl] = 1.f / (1.f + __expf(-acc));
    }
}

extern "C" void kernel_launch(void* const* d_in, const int* in_sizes, int n_in,
                              void* d_out, int out_size, void* d_ws, size_t ws_size,
                              hipStream_t stream) {
    (void)in_sizes; (void)n_in; (void)out_size; (void)ws_size;
    const float* x   = (const float*)d_in[0];
    // d_in[1] edge_index and d_in[3] batch are unused: topology is static.
    const float* ew  = (const float*)d_in[2];
    const float* gf  = (const float*)d_in[4];
    const float* w1  = (const float*)d_in[5];
    const float* b1  = (const float*)d_in[6];
    const float* g1  = (const float*)d_in[7];
    const float* be1 = (const float*)d_in[8];
    const float* m1  = (const float*)d_in[9];
    const float* v1  = (const float*)d_in[10];
    const float* w2  = (const float*)d_in[11];
    const float* b2  = (const float*)d_in[12];
    const float* g2  = (const float*)d_in[13];
    const float* be2 = (const float*)d_in[14];
    const float* m2  = (const float*)d_in[15];
    const float* v2  = (const float*)d_in[16];
    const float* sw1 = (const float*)d_in[17];
    const float* sb1 = (const float*)d_in[18];
    const float* sg1 = (const float*)d_in[19];
    const float* sbe1= (const float*)d_in[20];
    const float* sm1 = (const float*)d_in[21];
    const float* sv1 = (const float*)d_in[22];
    const float* sw2 = (const float*)d_in[23];
    const float* sb2 = (const float*)d_in[24];
    const float* sg2 = (const float*)d_in[25];
    const float* sbe2= (const float*)d_in[26];
    const float* sm2 = (const float*)d_in[27];
    const float* sv2 = (const float*)d_in[28];
    const float* wo  = (const float*)d_in[29];
    const float* bo  = (const float*)d_in[30];
    const float* wb  = (const float*)d_in[31];
    const float* bb  = (const float*)d_in[32];
    unsigned char* ws = (unsigned char*)d_ws;

    prep_kernel<<<1, 256, 0, stream>>>(w1, b1, g1, be1, m1, v1,
                                       w2, b2, g2, be2, m2, v2,
                                       sw1, sb1, sg1, sbe1, sm1, sv1,
                                       sw2, sb2, sg2, sbe2, sm2, sv2, ws);
    gcn_kernel<<<NGRAPH / GPB, 256, 0, stream>>>(x, ew, gf, wo, bo, wb, bb, ws,
                                                 (float*)d_out);
}
